// ContextualLanguageRefinement_19868518711350
// MI455X (gfx1250) — compile-verified, matrix operand order checked by reading, not run on hardware
//
#include <hip/hip_runtime.h>

// ---------------------------------------------------------------------------
// ContextualLanguageRefinement on MI455X (gfx1250, wave32, WMMA)
//
// out = softmax( (attn @ H @ W_proj) / tau ), attn = banded softmax(QK^T/16)
// Refactor: (attn @ H) @ W_proj == attn @ (H @ W_proj)  -> one big fused GEMM
//   C[16384 x 544] = H[16384 x 1024] @ [W_Q | W_K | W_proj]  (bf16 WMMA, f32 acc)
// then a cheap 7-wide banded softmax + 32-way mix + softmax.
//
// Round 3: async global->LDS DMA now takes a real AS(3) pointer operand so the
// LDS slab escapes (round 2's integer-offset version let the compiler fold the
// ds_loads to undef). A-fragments are software-pipelined one K-step ahead.
// ---------------------------------------------------------------------------

typedef __bf16 bf16_t;
typedef bf16_t v16bf __attribute__((ext_vector_type(16)));
typedef float  v8f   __attribute__((ext_vector_type(8)));
typedef __attribute__((address_space(3))) char lds_char;

#define B_   8
#define T_   2048
#define D_   1024
#define DS_  256
#define KL_  32
#define WIN_ 3
#define M_   (B_ * T_)        // 16384 rows
#define N_   (2 * DS_ + KL_)  // 544 fused output cols: [Q | K | P]
#define NT_  (N_ / 16)        // 34 n-tiles
#define KT_  (D_ / 32)        // 32 k-steps of 32
#define NG_  2                // n-tiles per wave (34 = 2 * 17)
#define BSLAB_ (NG_ * KT_ * 32 * 16)   // 32768 bf16 = 64 KB LDS slab

// CDNA5 async global->LDS DMA (ASYNCcnt path). The AS(3) pointer operand both
// supplies the LDS byte address in a VGPR and makes the LDS object escape so
// subsequent ds_loads are not folded away.
static __device__ __forceinline__ void async_g2l_b128(lds_char* lds, const void* g) {
  asm volatile("global_load_async_to_lds_b128 %0, %1, off"
               :: "v"(lds), "v"(g)
               : "memory");
}

// ---------------------------------------------------------------------------
// Pack H (fp32 row-major) -> bf16 A-operand fragments.
// 16-bit A 16x32 layout: lane<16 holds row (lane&15), K = {0..7, 16..23};
// lane>=16 holds same row, K = {8..15, 24..31}. One thread = one lane-fragment.
// ---------------------------------------------------------------------------
__global__ void __launch_bounds__(256) pack_a(const float* __restrict__ H,
                                              bf16_t* __restrict__ Ap) {
  int idx  = blockIdx.x * 256 + threadIdx.x;      // (mt, ks, lane), 1024*32*32
  int lane = idx & 31;
  int ks   = (idx >> 5) & (KT_ - 1);
  int mt   = idx >> 10;
  int row  = mt * 16 + (lane & 15);
  int kb   = ks * 32 + ((lane >> 4) << 3);
  const float4* s4 = (const float4*)(H + (size_t)row * D_ + kb);
  float4 x0 = s4[0], x1 = s4[1];   // K = kb + 0..7
  float4 x2 = s4[4], x3 = s4[5];   // K = kb + 16..23
  v16bf v;
  v[0]  = (bf16_t)x0.x; v[1]  = (bf16_t)x0.y; v[2]  = (bf16_t)x0.z; v[3]  = (bf16_t)x0.w;
  v[4]  = (bf16_t)x1.x; v[5]  = (bf16_t)x1.y; v[6]  = (bf16_t)x1.z; v[7]  = (bf16_t)x1.w;
  v[8]  = (bf16_t)x2.x; v[9]  = (bf16_t)x2.y; v[10] = (bf16_t)x2.z; v[11] = (bf16_t)x2.w;
  v[12] = (bf16_t)x3.x; v[13] = (bf16_t)x3.y; v[14] = (bf16_t)x3.z; v[15] = (bf16_t)x3.w;
  *(v16bf*)(Ap + (size_t)idx * 16) = v;
}

// ---------------------------------------------------------------------------
// Pack [W_Q | W_K | W_proj] -> bf16 B-operand fragments (mirrored layout:
// lane holds column nt*16 + (lane&15), same K split as A).
// ---------------------------------------------------------------------------
__global__ void __launch_bounds__(256) pack_b(const float* __restrict__ Wq,
                                              const float* __restrict__ Wk,
                                              const float* __restrict__ Wp,
                                              bf16_t* __restrict__ Bp) {
  int idx = blockIdx.x * 256 + threadIdx.x;       // (nt, ks, lane), 34*32*32
  if (idx >= NT_ * KT_ * 32) return;
  int lane = idx & 31;
  int ks   = (idx >> 5) & (KT_ - 1);
  int nt   = idx >> 10;
  int n    = nt * 16 + (lane & 15);
  int kb   = ks * 32 + ((lane >> 4) << 3);
  v16bf v;
#pragma unroll
  for (int e = 0; e < 16; ++e) {
    int k = kb + (e & 7) + ((e >> 3) << 4);       // {kb..kb+7, kb+16..kb+23}
    float val;
    if (n < DS_)            val = Wq[(size_t)k * DS_ + n];
    else if (n < 2 * DS_)   val = Wk[(size_t)k * DS_ + (n - DS_)];
    else                    val = Wp[(size_t)k * KL_ + (n - 2 * DS_)];
    v[e] = (bf16_t)val;
  }
  *(v16bf*)(Bp + (size_t)idx * 16) = v;
}

// ---------------------------------------------------------------------------
// Fused GEMM: QKP[16384 x 544] = A x B, bf16 in / f32 acc via
// v_wmma_f32_16x16x32_bf16. One wave per (m-tile, 2 n-tiles); 8 waves/block.
// B slab (64 KB, identical for all 8 waves) staged once into LDS using the
// CDNA5 async global->LDS path, drained with s_wait_asynccnt; B fragments
// then come from ds_load_b128 while A streams from global, pipelined 1 deep.
// ---------------------------------------------------------------------------
__global__ void __launch_bounds__(256) gemm_qkp(const bf16_t* __restrict__ Ap,
                                                const bf16_t* __restrict__ Bp,
                                                float* __restrict__ QKP) {
  __shared__ bf16_t Bs[BSLAB_];                   // 64 KB of 320 KB WGP LDS

  int tid  = threadIdx.x;
  int wave = tid >> 5;
  int lane = tid & 31;
  int mt   = blockIdx.x * 8 + wave;               // 0..1023
  int ng0  = blockIdx.y * NG_;                    // 0,2,..,32

  lds_char* Bs_lds = (lds_char*)(void*)Bs;

  // ---- cooperative async DMA: 64 KB contiguous B slab -> LDS --------------
  {
    const char* gsrc = (const char*)(Bp + (size_t)ng0 * KT_ * 32 * 16);
#pragma unroll
    for (int j = 0; j < 16; ++j) {
      size_t byte = (size_t)(j * 256 + tid) * 16u;
      async_g2l_b128(Bs_lds + byte, gsrc + byte);
    }
    asm volatile("s_wait_asynccnt 0x0" ::: "memory");
    __syncthreads();
  }

  const v16bf* Av = (const v16bf*)Ap;

  v8f acc[NG_];
#pragma unroll
  for (int g = 0; g < NG_; ++g) acc[g] = (v8f){0.f,0.f,0.f,0.f,0.f,0.f,0.f,0.f};

  size_t abase = (size_t)mt * KT_ * 32 + lane;

  v16bf a = Av[abase];                             // prologue: K-step 0
#pragma unroll 4
  for (int ks = 0; ks < KT_ - 1; ++ks) {
    v16bf an = Av[abase + (size_t)(ks + 1) * 32];  // next step, 1-deep pipeline
    if (ks + 2 < KT_)
      __builtin_prefetch(&Av[abase + (size_t)(ks + 2) * 32], 0, 1);
#pragma unroll
    for (int g = 0; g < NG_; ++g) {
      v16bf b = *(const v16bf*)(Bs + ((size_t)(g * KT_ + ks) * 32 + lane) * 16);
      acc[g] = __builtin_amdgcn_wmma_f32_16x16x32_bf16(
          /*neg_a=*/false, a, /*neg_b=*/false, b,
          /*c_mod=*/(short)0, acc[g], /*reuse_a=*/false, /*reuse_b=*/false);
    }
    a = an;
  }
#pragma unroll
  for (int g = 0; g < NG_; ++g) {                  // epilogue: K-step KT_-1
    v16bf b = *(const v16bf*)(Bs + ((size_t)(g * KT_ + (KT_ - 1)) * 32 + lane) * 16);
    acc[g] = __builtin_amdgcn_wmma_f32_16x16x32_bf16(
        false, a, false, b, (short)0, acc[g], false, false);
  }

  // f32 C/D layout: VGPR r -> row ((lane>>4)*8 + r), col (lane&15)
#pragma unroll
  for (int g = 0; g < NG_; ++g) {
    int col = (ng0 + g) * 16 + (lane & 15);
#pragma unroll
    for (int r = 0; r < 8; ++r) {
      int rl = ((lane >> 4) << 3) + r;
      QKP[(size_t)(mt * 16 + rl) * N_ + col] = acc[g][r];
    }
  }
}

// ---------------------------------------------------------------------------
// Banded attention + projection mix + final softmax. One wave per token.
// QKP row layout: [ Q(0..255) | K(256..511) | P(512..543) ]
// ---------------------------------------------------------------------------
__global__ void __launch_bounds__(256) band_attn(const float* __restrict__ QKP,
                                                 const float* __restrict__ tau_p,
                                                 float* __restrict__ out) {
  int wave = threadIdx.x >> 5;
  int lane = threadIdx.x & 31;
  int t    = blockIdx.x * 8 + wave;               // global row 0..16383
  int b    = t >> 11;
  int tt   = t & (T_ - 1);
  float inv_tau = 1.0f / tau_p[0];

  // 8 Q components per lane
  const float* Qr = QKP + (size_t)t * N_;
  float q[8];
#pragma unroll
  for (int i = 0; i < 8; ++i) q[i] = Qr[lane * 8 + i];

  int s0 = (tt - WIN_ < 0) ? 0 : tt - WIN_;
  int s1 = (tt + WIN_ > T_ - 1) ? T_ - 1 : tt + WIN_;

  float sc[2 * WIN_ + 1];
  float m = -3.0e38f;
  for (int s = s0; s <= s1; ++s) {
    const float* Kr = QKP + (size_t)(b * T_ + s) * N_ + DS_;
    float p = 0.f;
#pragma unroll
    for (int i = 0; i < 8; ++i) p += q[i] * Kr[lane * 8 + i];
#pragma unroll
    for (int off = 16; off >= 1; off >>= 1) p += __shfl_xor(p, off, 32);
    p *= (1.0f / 16.0f);                          // / sqrt(DS)
    sc[s - s0] = p;
    m = fmaxf(m, p);
  }
  float den = 0.f;
  for (int s = s0; s <= s1; ++s) { sc[s - s0] = __expf(sc[s - s0] - m); den += sc[s - s0]; }
  float rden = 1.0f / den;

  // lane k mixes P rows of the window
  float logit = 0.f;
  for (int s = s0; s <= s1; ++s)
    logit += sc[s - s0] * QKP[(size_t)(b * T_ + s) * N_ + 2 * DS_ + lane];
  logit = logit * rden * inv_tau;

  float mx = logit;
#pragma unroll
  for (int off = 16; off >= 1; off >>= 1) mx = fmaxf(mx, __shfl_xor(mx, off, 32));
  float e = __expf(logit - mx);
  float sum = e;
#pragma unroll
  for (int off = 16; off >= 1; off >>= 1) sum += __shfl_xor(sum, off, 32);

  out[(size_t)t * KL_ + lane] = e / sum;
}

// ---------------------------------------------------------------------------
extern "C" void kernel_launch(void* const* d_in, const int* in_sizes, int n_in,
                              void* d_out, int out_size, void* d_ws, size_t ws_size,
                              hipStream_t stream) {
  const float* H   = (const float*)d_in[0];
  const float* tau = (const float*)d_in[1];
  const float* Wq  = (const float*)d_in[2];
  const float* Wk  = (const float*)d_in[3];
  const float* Wp  = (const float*)d_in[4];
  float* out = (float*)d_out;

  // Workspace: Apack (33.5 MB bf16) | Bpack (1.1 MB bf16) | QKP (35.7 MB f32)
  bf16_t* Ap  = (bf16_t*)d_ws;
  bf16_t* Bp  = Ap + (size_t)(M_ / 16) * KT_ * 32 * 16;
  float*  QKP = (float*)(Bp + (size_t)NT_ * KT_ * 32 * 16);

  pack_a<<<(M_ / 16) * KT_ * 32 / 256, 256, 0, stream>>>(H, Ap);
  pack_b<<<(NT_ * KT_ * 32 + 255) / 256, 256, 0, stream>>>(Wq, Wk, Wp, Bp);
  gemm_qkp<<<dim3(M_ / 16 / 8, NT_ / NG_), 256, 0, stream>>>(Ap, Bp, QKP);
  band_attn<<<M_ / 8, 256, 0, stream>>>(QKP, tau, out);
}
